// WaveOperator2D_40381282517277
// MI455X (gfx1250) — compile-verified, matrix operand order checked by reading, not run on hardware
//
#include <hip/hip_runtime.h>
#include <math.h>

// ---------------------------------------------------------------------------
// WaveOperator2D for MI455X (gfx1250, wave32, WMMA bf16 16x16x32)
// B=8, C=128, H=W=128.  All contractions have K=128 -> 4 x wmma_16x16x32_bf16.
// All GEMM B operands are row-major [col][128] so staging is always b128 loads.
// ---------------------------------------------------------------------------

typedef __bf16 bf16t;
typedef __attribute__((ext_vector_type(16))) __bf16 v16bf;
typedef __attribute__((ext_vector_type(8)))  float  v8f;
typedef __attribute__((ext_vector_type(4)))  unsigned int v4u;

#define BB    8
#define CC    128
#define HHH   128
#define WPIX  128
#define HWPX  16384      // H*W
#define PLANE 16384      // one (b,c) plane
#define CHW   2097152    // C*H*W (per-batch elements)

#define MODE_INPROJ   0  // +bias; rows<128 -> carrier bf16 planar; rows>=128 -> silu(gate) f32 pixel-major
#define MODE_PHASE    1  // +bias, gelu, cos/sin modulation field -> f32 planar
#define MODE_TBF16    2  // transposed store bf16
#define MODE_MODBF16  3  // * modulation field, normal store bf16
#define MODE_NF32     4  // normal store f32
#define MODE_BIASNF32 5  // +bias, normal store f32 (out projection -> d_out planar)

__device__ __forceinline__ bf16t f2bf(float f) {
  union { float f; unsigned u; } a; a.f = f;
  unsigned r = (a.u + 0x7FFFu + ((a.u >> 16) & 1u)) >> 16;   // RNE
  union { unsigned short s; bf16t b; } o; o.s = (unsigned short)r;
  return o.b;
}

// ---------------------------------------------------------------------------
// WMMA GEMM: C[16 x 64] per wave, K = 128 fixed.
//   A: bf16 row-major [M x 128], lda elements, batched by sAz.
//   B: bf16 row-major [Ncols x 128] (element (k,n) = Bm[n*128 + k]), batch sBz.
//   B tile [64 n][32 k] staged in LDS per K-step; fragment reads are
//   ds_load_b128 pairs matching the 16-bit A/B VGPR layout (ISA 7.12.2).
// ---------------------------------------------------------------------------
__global__ __launch_bounds__(32)
void wo_wmma_gemm(const bf16t* __restrict__ A, int lda, long sAz,
                  const bf16t* __restrict__ Bm, long sBz,
                  void* __restrict__ outv, int ldo, long sOz,
                  void* __restrict__ out2v,
                  const float* __restrict__ bias,
                  const float* __restrict__ aux,
                  const float* __restrict__ sc0,
                  const float* __restrict__ sc1,
                  int mode)
{
  __shared__ __align__(16) bf16t ldsB[64][32];

  const int lane = threadIdx.x;
  const int n0 = blockIdx.x * 64;
  const int m0 = blockIdx.y * 16;
  const int z  = blockIdx.z;

  const bf16t* Ab = A  + (long)z * sAz;
  const bf16t* Bb = Bm + (long)z * sBz;

  const int cl = lane & 15;           // column / row-in-tile selector
  const int hi = lane >> 4;           // lane-half
  const int kb = hi ? 8 : 0;          // K base for this half (ISA layout)
  const int mrow = m0 + cl;

  v8f acc[4];
  v8f zero = {0.f,0.f,0.f,0.f,0.f,0.f,0.f,0.f};
  acc[0] = zero; acc[1] = zero; acc[2] = zero; acc[3] = zero;

  for (int s = 0; s < 4; ++s) {
    const int ks = s * 32;

    // ---- stage B tile [64 n][32 k]: 2 rows per lane, b128 loads ----
    #pragma unroll
    for (int r = 0; r < 2; ++r) {
      const int n = 2 * lane + r;
      const bf16t* src = Bb + (long)(n0 + n) * 128 + ks;
      v4u* dst = (v4u*)(&ldsB[n][0]);
      dst[0] = ((const v4u*)src)[0];      // 16B
      dst[1] = ((const v4u*)src)[1];      // 16B
    }
    __syncthreads();

    // ---- A fragment: two contiguous 16B runs per lane ----
    union { v16bf v; v4u d[2]; } af;
    const bf16t* ap = Ab + (long)mrow * lda + ks + kb;
    af.d[0] = *(const v4u*)(ap);            // K = ks+kb+0..7
    af.d[1] = *(const v4u*)(ap + 16);       // K = ks+kb+16..23
    if (s < 3) __builtin_prefetch((const void*)(ap + 32), 0, 0);

    // ---- 4 N-subtiles share the A fragment ----
    #pragma unroll
    for (int j = 0; j < 4; ++j) {
      union { v16bf v; v4u d[2]; } bfj;
      const bf16t* bp = &ldsB[j * 16 + cl][kb];
      bfj.d[0] = *(const v4u*)(bp);
      bfj.d[1] = *(const v4u*)(bp + 16);
      acc[j] = __builtin_amdgcn_wmma_f32_16x16x32_bf16(
          false, af.v, false, bfj.v, (short)0, acc[j], false, false);
    }
    __syncthreads();
  }

  // ---- epilogue: C frag element r maps to (row = m0 + 8*hi + r, col) ----
  const int rowbase = m0 + 8 * hi;

  #pragma unroll
  for (int j = 0; j < 4; ++j) {
    const int col = n0 + j * 16 + cl;

    if (mode == MODE_TBF16) {
      bf16t* ob = (bf16t*)outv + (long)z * sOz;
      union { v4u d; bf16t h[8]; } pk;
      #pragma unroll
      for (int r = 0; r < 8; ++r) pk.h[r] = f2bf(acc[j][r]);
      *(v4u*)(ob + (long)col * ldo + rowbase) = pk.d;       // contiguous 16B
    } else if (mode == MODE_MODBF16) {
      bf16t* ob = (bf16t*)outv + (long)z * sOz;
      const float* mp = aux + (long)(z % CC) * PLANE + (long)col * WPIX + rowbase;
      #pragma unroll
      for (int r = 0; r < 8; ++r)
        ob[(long)(rowbase + r) * ldo + col] = f2bf(acc[j][r] * mp[r]);
    } else if (mode == MODE_NF32) {
      float* ob = (float*)outv + (long)z * sOz;
      #pragma unroll
      for (int r = 0; r < 8; ++r)
        ob[(long)(rowbase + r) * ldo + col] = acc[j][r];
    } else if (mode == MODE_BIASNF32) {
      float* ob = (float*)outv + (long)z * sOz;
      #pragma unroll
      for (int r = 0; r < 8; ++r)
        ob[(long)(rowbase + r) * ldo + col] = acc[j][r] + bias[rowbase + r];
    } else if (mode == MODE_PHASE) {
      float* ob = (float*)outv;
      const float wsv = *sc0;
      const float dpv = *sc1;
      const float inv = 1.f / fmaxf(fabsf(wsv), 1e-6f);
      #pragma unroll
      for (int r = 0; r < 8; ++r) {
        float v  = acc[j][r] + bias[rowbase + r];
        float g  = 0.5f * v * (1.f + erff(v * 0.70710678118f));   // exact gelu
        float wp = wsv * g;
        float mv = cosf(wp) + (sinf(wp) * inv) * (1.f + 0.5f * dpv);
        ob[(long)(rowbase + r) * ldo + col] = mv;
      }
    } else { // MODE_INPROJ
      if (m0 < CC) {   // carrier rows -> bf16 channel-planar [c][pixel]
        bf16t* ob = (bf16t*)outv + (long)z * sOz;
        #pragma unroll
        for (int r = 0; r < 8; ++r)
          ob[(long)(rowbase + r) * ldo + col] = f2bf(acc[j][r] + bias[rowbase + r]);
      } else {         // gate rows -> silu, f32 pixel-major [pixel][c]
        float* og = (float*)out2v + (long)z * sOz;
        #pragma unroll
        for (int r = 0; r < 8; ++r) {
          float v = acc[j][r] + bias[rowbase + r];
          og[(long)col * CC + (rowbase - CC) + r] = v / (1.f + expf(-v));
        }
      }
    }
  }
}

// ---------------------------------------------------------------------------
// Depthwise 3x3 conv: planar f32 in -> PIXEL-MAJOR bf16 out ([pixel][c]),
// transposed through LDS so both reads and writes are coalesced.
// Block: one h-row x 16 channels; 256 threads = 2 c-slices x 128 w.
// ---------------------------------------------------------------------------
__global__ __launch_bounds__(256)
void wo_dwconv(const float* __restrict__ x, const float* __restrict__ w,
               const float* __restrict__ b, bf16t* __restrict__ out)
{
  __shared__ float tile[16][128];           // [c][w]
  const int h  = blockIdx.x;
  const int c0 = blockIdx.y * 16;
  const int bz = blockIdx.z;
  const int tid  = threadIdx.x;
  const int wq   = tid & 127;
  const int csub = tid >> 7;                // 0..1

  for (int cc = csub; cc < 16; cc += 2) {
    const int c = c0 + cc;
    const float* xp = x + ((long)bz * CC + c) * (long)HWPX;
    float wk[9];
    #pragma unroll
    for (int i = 0; i < 9; ++i) wk[i] = w[c * 9 + i];
    float acc = b[c];
    #pragma unroll
    for (int dh = -1; dh <= 1; ++dh) {
      const int hh = h + dh;
      if (hh < 0 || hh >= HHH) continue;
      #pragma unroll
      for (int dw = -1; dw <= 1; ++dw) {
        const int ww = wq + dw;
        if (ww < 0 || ww >= WPIX) continue;
        acc += xp[hh * WPIX + ww] * wk[(dh + 1) * 3 + (dw + 1)];
      }
    }
    tile[cc][wq] = acc;
  }
  __syncthreads();

  bf16t* ob = out + (long)bz * CHW + (long)h * WPIX * CC;   // row h, pixel-major
  for (int t = tid; t < 16 * 128; t += 256) {
    const int c  = t & 15;                  // adjacent threads -> adjacent c
    const int wo = t >> 4;
    ob[(long)wo * CC + c0 + c] = f2bf(tile[c][wo]);
  }
}

// ---------------------------------------------------------------------------
// DCT-II basis in bf16 (and its transpose).
// ---------------------------------------------------------------------------
__global__ __launch_bounds__(128)
void wo_basis(bf16t* __restrict__ cb, bf16t* __restrict__ cbt)
{
  const int h = threadIdx.x;
  const int n = blockIdx.x;
  float v = cosf((float)n * ((h + 0.5f) / 128.f) * 3.14159265358979f)
            * 0.125f * 1.41421356237f;            // sqrt(2/64)
  if (n == 0) v *= 0.70710678118f;
  cb [n * 128 + h] = f2bf(v);
  cbt[h * 128 + n] = f2bf(v);
}

__global__ void wo_cvt_bf16(const float* __restrict__ in, bf16t* __restrict__ out, int n)
{
  for (int i = blockIdx.x * blockDim.x + threadIdx.x; i < n; i += gridDim.x * blockDim.x)
    out[i] = f2bf(in[i]);
}

// ---------------------------------------------------------------------------
// LayerNorm(C) over planar fused + gate multiply -> bf16 pixel-major.
// 32-pixel x 128-channel tile, LDS transpose.
// ---------------------------------------------------------------------------
__global__ __launch_bounds__(256)
void wo_ln_gate(const float* __restrict__ fused, const float* __restrict__ gate,
                const float* __restrict__ lw, const float* __restrict__ lb,
                bf16t* __restrict__ out)
{
  __shared__ float tile[32][129];
  __shared__ float mu[32], rs[32];

  const int bz = blockIdx.y;
  const int p0 = blockIdx.x * 32;
  const float* fb = fused + (long)bz * CHW;
  const float* gb = gate  + (long)bz * CHW;
  bf16t*       ob = out   + (long)bz * CHW;
  const int tid = threadIdx.x;

  for (int t = tid; t < 32 * 128; t += 256) {
    const int pl = t & 31;
    const int c  = t >> 5;
    tile[pl][c] = fb[(long)c * HWPX + p0 + pl];     // coalesced planar read
  }
  __syncthreads();

  if (tid < 32) {
    float s = 0.f, s2 = 0.f;
    #pragma unroll 4
    for (int c = 0; c < 128; ++c) { float v = tile[tid][c]; s += v; s2 += v * v; }
    const float m = s * (1.f / 128.f);
    mu[tid] = m;
    rs[tid] = rsqrtf(s2 * (1.f / 128.f) - m * m + 1e-5f);
  }
  __syncthreads();

  for (int t = tid; t < 32 * 128; t += 256) {
    const int c  = t & 127;
    const int pl = t >> 7;
    const float v = (tile[pl][c] - mu[pl]) * rs[pl] * lw[c] + lb[c];
    const long idx = (long)(p0 + pl) * CC + c;
    ob[idx] = f2bf(v * gb[idx]);                    // gate already silu'd
  }
}

// ---------------------------------------------------------------------------
extern "C" void kernel_launch(void* const* d_in, const int* in_sizes, int n_in,
                              void* d_out, int out_size, void* d_ws, size_t ws_size,
                              hipStream_t stream)
{
  const float* x     = (const float*)d_in[0];
  const float* dw_w  = (const float*)d_in[1];
  const float* dw_b  = (const float*)d_in[2];
  const float* in_w  = (const float*)d_in[3];
  const float* in_b  = (const float*)d_in[4];
  const float* out_w = (const float*)d_in[5];
  const float* out_b = (const float*)d_in[6];
  const float* ln_w  = (const float*)d_in[7];
  const float* ln_b  = (const float*)d_in[8];
  const float* ph_w  = (const float*)d_in[9];
  const float* ph_b  = (const float*)d_in[10];
  const float* wspd  = (const float*)d_in[11];
  const float* damp  = (const float*)d_in[12];
  const float* femb  = (const float*)d_in[13];
  float* outp = (float*)d_out;

  char* ws = (char*)d_ws;
  // workspace layout (fused f32 region reuses mixed+carrier region: both dead
  // before the iDCT writes fused; lnout reuses the S2 region).
  float* fused   = (float*)(ws + 0);               // 64MB [B][C][H][W]
  bf16t* mixed   = (bf16t*)(ws + 0);               // 32MB pixel-major (dead after in_proj)
  bf16t* carrier = (bf16t*)(ws + 33554432);        // 32MB planar (dead after S1)
  float* gate    = (float*)(ws + 67108864);        // 64MB [B][HW][C] (silu'd)
  bf16t* Tbuf    = (bf16t*)(ws + 134217728);       // 32MB (T1, reused as T2)
  bf16t* S2buf   = (bf16t*)(ws + 167772160);       // 32MB (reused as lnout)
  bf16t* lnout   = S2buf;
  float* modp    = (float*)(ws + 201326592);       // 8MB  [C][H][W]
  bf16t* febf    = (bf16t*)(ws + 209715200);       // 4MB
  bf16t* inw_bf  = (bf16t*)(ws + 213909504);       // 64KB
  bf16t* phw_bf  = (bf16t*)(ws + 213975040);       // 32KB
  bf16t* outw_bf = (bf16t*)(ws + 214007808);       // 32KB
  bf16t* cosb    = (bf16t*)(ws + 214040576);       // 32KB
  bf16t* cosbT   = (bf16t*)(ws + 214073344);       // 32KB
  (void)in_sizes; (void)n_in; (void)out_size; (void)ws_size;

  // small precomputation
  wo_basis<<<dim3(128), dim3(128), 0, stream>>>(cosb, cosbT);
  wo_cvt_bf16<<<dim3(64),   dim3(256), 0, stream>>>(in_w,  inw_bf, 32768);
  wo_cvt_bf16<<<dim3(32),   dim3(256), 0, stream>>>(ph_w,  phw_bf, 16384);
  wo_cvt_bf16<<<dim3(32),   dim3(256), 0, stream>>>(out_w, outw_bf, 16384);
  wo_cvt_bf16<<<dim3(4096), dim3(256), 0, stream>>>(femb,  febf,  HWPX * CC);

  // depthwise conv -> mixed bf16 PIXEL-MAJOR [b][pixel][c]
  wo_dwconv<<<dim3(HHH, CC / 16, BB), dim3(256), 0, stream>>>(x, dw_w, dw_b, mixed);

  // modulation field (batch-independent): modp[c][h][w]
  wo_wmma_gemm<<<dim3(256, 8, 1), dim3(32), 0, stream>>>(
      phw_bf, 128, 0,  febf, 0,
      modp, HWPX, 0, nullptr, ph_b, nullptr, wspd, damp, MODE_PHASE);

  // in-projection (swapped): in_w[256x128] @ mixed^T -> carrier + silu(gate)
  wo_wmma_gemm<<<dim3(256, 16, BB), dim3(32), 0, stream>>>(
      inw_bf, 128, 0,  mixed, (long)CHW,
      carrier, HWPX, (long)CHW, gate, in_b, nullptr, nullptr, nullptr, MODE_INPROJ);

  // S1 = X @ C^T  (per (b,c) plane), transposed-store -> T1[m][h]
  wo_wmma_gemm<<<dim3(2, 8, BB * CC), dim3(32), 0, stream>>>(
      carrier, 128, PLANE,  cosb, 0,
      Tbuf, 128, PLANE, nullptr, nullptr, nullptr, nullptr, nullptr, MODE_TBF16);

  // S2 = T1 @ C^T, fused * mod[n][m] -> S2mod[m][n]
  wo_wmma_gemm<<<dim3(2, 8, BB * CC), dim3(32), 0, stream>>>(
      Tbuf, 128, PLANE,  cosb, 0,
      S2buf, 128, PLANE, nullptr, nullptr, modp, nullptr, nullptr, MODE_MODBF16);

  // G1 = S2mod @ C, transposed-store -> T2[h][m]
  wo_wmma_gemm<<<dim3(2, 8, BB * CC), dim3(32), 0, stream>>>(
      S2buf, 128, PLANE,  cosbT, 0,
      Tbuf, 128, PLANE, nullptr, nullptr, nullptr, nullptr, nullptr, MODE_TBF16);

  // G2 = T2 @ C -> fused f32 planar
  wo_wmma_gemm<<<dim3(2, 8, BB * CC), dim3(32), 0, stream>>>(
      Tbuf, 128, PLANE,  cosbT, 0,
      fused, 128, PLANE, nullptr, nullptr, nullptr, nullptr, nullptr, MODE_NF32);

  // LayerNorm + gate -> bf16 pixel-major
  wo_ln_gate<<<dim3(HWPX / 32, BB), dim3(256), 0, stream>>>(
      fused, gate, ln_w, ln_b, lnout);

  // out-projection (swapped): out_w @ lnout^T -> d_out planar [B][C][H][W]
  wo_wmma_gemm<<<dim3(256, 8, BB), dim3(32), 0, stream>>>(
      outw_bf, 128, 0,  lnout, (long)CHW,
      outp, HWPX, (long)CHW, nullptr, out_b, nullptr, nullptr, nullptr, MODE_BIASNF32);
}
